// AIMv2Attention_71382356459767
// MI455X (gfx1250) — matrix operand order repaired and checked by reading, not compiled, not run on hardware
//
#include <hip/hip_runtime.h>
#include <hip/hip_bf16.h>

// ---------------------------------------------------------------------------
// AIMv2 attention on gfx1250: bf16 WMMA pipeline, f32 accumulation,
// LDS-staged tiles (double buffered, async-copy when available),
// exp2-domain online softmax with matrix-pipe row sums.
// B=4, S=2048, D=1024, H=16, d=64.
// ---------------------------------------------------------------------------

typedef __attribute__((ext_vector_type(16))) __bf16 v16bf;
typedef __attribute__((ext_vector_type(8)))  float  v8f;
typedef int v4i __attribute__((__vector_size__(16)));

#ifndef __has_builtin
#define __has_builtin(x) 0
#endif
#if __has_builtin(__builtin_amdgcn_global_load_async_to_lds_b128)
#define HAVE_ASYNC_LDS 1
#else
#define HAVE_ASYNC_LDS 0
#endif

union FragU {
    v16bf v;
    uint4 q[2];
};

// A-operand (16-bit, 16x32): elements 0..15 -> K = kbase+e
// (kbase = 16*(lane>=16)): 32 contiguous bytes at p.
__device__ __forceinline__ v16bf load_frag_a(const __bf16* p) {
    FragU f;
    f.q[0] = *(const uint4*)(p);
    f.q[1] = *(const uint4*)(p + 8);
    return f.v;
}

// B-operand (16-bit, 32x16): lanes 0..15 hold K=0..15, lanes 16..31 K=16..31,
// N = lane&15. K contiguous in memory -> 32 contiguous bytes at p.
__device__ __forceinline__ v16bf load_frag_b(const __bf16* p) {
    FragU f;
    f.q[0] = *(const uint4*)(p);
    f.q[1] = *(const uint4*)(p + 8);
    return f.v;
}

__device__ __forceinline__ v16bf ones_frag() {
    FragU f;
    uint4 o;
    o.x = o.y = o.z = o.w = 0x3f803f80u;   // bf16 1.0 pair
    f.q[0] = o;
    f.q[1] = o;
    return f.v;
}

__device__ __forceinline__ v8f wmma_bf16(v16bf a, v16bf b, v8f c) {
    return __builtin_amdgcn_wmma_f32_16x16x32_bf16(
        false, a, false, b, (short)0, c, false, false);
}

// 16B global -> LDS copy: async (ASYNCcnt) when the builtin exists, else
// a VGPR round-trip (load + ds_store).
__device__ __forceinline__ void async_copy_b128(const __bf16* g, __bf16* l) {
#if HAVE_ASYNC_LDS
    __builtin_amdgcn_global_load_async_to_lds_b128(
        (v4i __attribute__((address_space(1)))*)(g),
        (v4i __attribute__((address_space(3)))*)(l),
        0, 0);
#else
    *(uint4*)l = *(const uint4*)g;
#endif
}

__device__ __forceinline__ void async_copy_wait() {
#if HAVE_ASYNC_LDS
    asm volatile("s_wait_asynccnt 0x0" ::: "memory");
#endif
}

// ---------------------------------------------------------------------------
// fp32 -> bf16 conversions / transposes
// ---------------------------------------------------------------------------
__global__ void cvt_f32_bf16(const float* __restrict__ in,
                             __bf16* __restrict__ out, int n) {
    int i = blockIdx.x * blockDim.x + threadIdx.x;
    if (i < n) out[i] = (__bf16)in[i];
}

__global__ void transpose_cvt(const float* __restrict__ in,
                              __bf16* __restrict__ out, int rows, int cols) {
    int c = blockIdx.x * blockDim.x + threadIdx.x;
    int r = blockIdx.y;
    if (c < cols) out[(size_t)c * rows + r] = (__bf16)in[(size_t)r * cols + c];
}

// qkv bf16 [B*S, 3072], V at col 2048 + h*64 + dd -> vt [b][h][dd][s]
__global__ void transpose_v(const __bf16* __restrict__ qkv,
                            __bf16* __restrict__ vt) {
    int i = blockIdx.x * blockDim.x + threadIdx.x;
    int s  = i & 2047;
    int dd = (i >> 11) & 63;
    int h  = (i >> 17) & 15;
    int b  = i >> 21;
    vt[i] = qkv[(b * 2048 + s) * 3072 + 2048 + h * 64 + dd];
}

// ---------------------------------------------------------------------------
// WMMA GEMM with LDS double-buffered staging (async copies when available).
// C[M,N] = A[M,K] * W[K,N] (W given transposed: BT[N,K]).
// Block = 256 threads = 8 waves; block tile 128x128; wave tile 64x32;
// K-step 32. LDS: 2*(8KB A + 8KB B) = 32 KB.
// Columns n < qcols are scaled by qscale before the bf16 store (folds the
// attention softmax scale into Q).
// ---------------------------------------------------------------------------
template <bool STORE_BF16>
__global__ __launch_bounds__(256, 1) void gemm_wmma(
        const __bf16* __restrict__ A, const __bf16* __restrict__ BT,
        float* __restrict__ Cf, __bf16* __restrict__ Cb,
        int M, int N, int K, int qcols, float qscale) {
    __shared__ __bf16 at[2][128 * 32];
    __shared__ __bf16 bt[2][128 * 32];

    const int tid  = threadIdx.x;
    const int lane = tid & 31;
    const int wid  = tid >> 5;
    const int half = lane >> 4;
    const int r    = lane & 15;
    const int m0b = blockIdx.y * 128;
    const int n0b = blockIdx.x * 128;
    const int mw = (wid >> 2) * 64;
    const int nw = (wid & 3) * 32;

    // staging units: 512 uint4 per operand tile; this thread owns u and u+256.
    // unit u -> row u/4, k-offset (u&3)*8, LDS element 8u.
    const int u0 = tid, u1 = tid + 256;
    const int ar0 = (m0b + (u0 >> 2)) * K + ((u0 & 3) << 3);
    const int ar1 = (m0b + (u1 >> 2)) * K + ((u1 & 3) << 3);
    const int br0 = (n0b + (u0 >> 2)) * K + ((u0 & 3) << 3);
    const int br1 = (n0b + (u1 >> 2)) * K + ((u1 & 3) << 3);

    // prologue: stage k0 = 0 into buffer 0
    async_copy_b128(A + ar0, &at[0][u0 * 8]);
    async_copy_b128(A + ar1, &at[0][u1 * 8]);
    async_copy_b128(BT + br0, &bt[0][u0 * 8]);
    async_copy_b128(BT + br1, &bt[0][u1 * 8]);

    v8f acc[4][2];
#pragma unroll
    for (int i = 0; i < 4; i++)
#pragma unroll
        for (int j = 0; j < 2; j++) acc[i][j] = (v8f){};

    int p = 0;
    for (int k0 = 0; k0 < K; k0 += 32) {
        async_copy_wait();      // this wave's staged writes are in LDS
        __syncthreads();        // buffer p is fully staged by all waves

        if (k0 + 32 < K) {      // stage next slab; overlaps with compute below
            async_copy_b128(A + ar0 + k0 + 32, &at[p ^ 1][u0 * 8]);
            async_copy_b128(A + ar1 + k0 + 32, &at[p ^ 1][u1 * 8]);
            async_copy_b128(BT + br0 + k0 + 32, &bt[p ^ 1][u0 * 8]);
            async_copy_b128(BT + br1 + k0 + 32, &bt[p ^ 1][u1 * 8]);
        }

        v16bf a[4], b[2];
#pragma unroll
        for (int i = 0; i < 4; i++)
            a[i] = load_frag_a(&at[p][(mw + 16 * i + r) * 32 + half * 16]);
#pragma unroll
        for (int j = 0; j < 2; j++)
            b[j] = load_frag_b(&bt[p][(nw + 16 * j + r) * 32 + half * 16]);
#pragma unroll
        for (int i = 0; i < 4; i++)
#pragma unroll
            for (int j = 0; j < 2; j++)
                acc[i][j] = wmma_bf16(a[i], b[j], acc[i][j]);

        p ^= 1;
    }

#pragma unroll
    for (int i = 0; i < 4; i++)
#pragma unroll
        for (int j = 0; j < 2; j++) {
            const int n_tile = n0b + nw + 16 * j;
            const bool isq = n_tile < qcols;
#pragma unroll
            for (int v = 0; v < 8; v++) {
                int m = m0b + mw + 16 * i + v + 8 * half;
                int n = n_tile + r;
                float val = acc[i][j][v];
                if (STORE_BF16) {
                    if (isq) val *= qscale;
                    Cb[m * N + n] = (__bf16)val;
                } else {
                    Cf[m * N + n] = val;
                }
            }
        }
}

// ---------------------------------------------------------------------------
// Flash attention. Block = 4 waves = 64 queries of one (b,h). K/V chunk tiles
// (32 keys) staged cooperatively in LDS (async when available), double
// buffered, one barrier per chunk. Row sums via wmma against a ones B-operand
// (broadcast to all lanes); only the max needs lane shuffles. Q pre-scaled by
// 0.125*log2(e) -> scores are in the exp2 domain.
// ---------------------------------------------------------------------------
__global__ __launch_bounds__(128, 1) void attn_wmma(
        const __bf16* __restrict__ qkv, const __bf16* __restrict__ vt,
        __bf16* __restrict__ ctx) {
    constexpr int S = 2048;
    __shared__ __bf16 kt[2][32 * 64];     // [key][dd]
    __shared__ __bf16 vtile[2][64 * 32];  // [dd][key]
    __shared__ __bf16 plds[4][16 * 32];   // per-wave P staging

    const int tid  = threadIdx.x;
    const int lane = tid & 31;
    const int w    = tid >> 5;
    const int half = lane >> 4;
    const int r    = lane & 15;

    const int bh = blockIdx.x >> 5;
    const int qt = blockIdx.x & 31;
    const int b  = bh >> 4;
    const int h  = bh & 15;
    const int q0 = qt * 64 + w * 16;

    const int base_q = b * S * 3072 + h * 64;
    const int base_k = base_q + 1024;
    const int base_v = bh * 64 * 2048;

    // staging: K tile unit u -> key u/8, dd (u&7)*8 ; V tile unit u -> dd u/4,
    // key (u&3)*8 ; LDS element = 8u. This thread owns u and u+128.
    const int ku0 = tid, ku1 = tid + 128;
    const int ksrc0 = base_k + (ku0 >> 3) * 3072 + ((ku0 & 7) << 3);
    const int ksrc1 = base_k + (ku1 >> 3) * 3072 + ((ku1 & 7) << 3);
    const int vsrc0 = base_v + (ku0 >> 2) * 2048 + ((ku0 & 3) << 3);
    const int vsrc1 = base_v + (ku1 >> 2) * 2048 + ((ku1 & 3) << 3);

    // prologue: stage chunk 0 into buffer 0
    async_copy_b128(qkv + ksrc0, &kt[0][ku0 * 8]);
    async_copy_b128(qkv + ksrc1, &kt[0][ku1 * 8]);
    async_copy_b128(vt + vsrc0, &vtile[0][ku0 * 8]);
    async_copy_b128(vt + vsrc1, &vtile[0][ku1 * 8]);

    // Q fragments for the two 32-wide d-chunks (pre-scaled, exp2 domain)
    v16bf aq[2];
#pragma unroll
    for (int c = 0; c < 2; c++)
        aq[c] = load_frag_a(qkv + base_q + (q0 + r) * 3072 + c * 32 + half * 16);

    const v16bf vones = ones_frag();

    float row_m[8];
#pragma unroll
    for (int v = 0; v < 8; v++) row_m[v] = -3.0e38f;

    v8f cacc[4];
#pragma unroll
    for (int t = 0; t < 4; t++) cacc[t] = (v8f){};
    v8f lacc = (v8f){};

    __bf16* pl = &plds[w][0];
    int p = 0;

    for (int kc = 0; kc < S; kc += 32) {
        async_copy_wait();
        __syncthreads();   // buffer p holds chunk `kc`

        if (kc + 32 < S) { // stage next chunk; overlaps with compute below
            async_copy_b128(qkv + ksrc0 + (kc + 32) * 3072, &kt[p ^ 1][ku0 * 8]);
            async_copy_b128(qkv + ksrc1 + (kc + 32) * 3072, &kt[p ^ 1][ku1 * 8]);
            async_copy_b128(vt + vsrc0 + kc + 32, &vtile[p ^ 1][ku0 * 8]);
            async_copy_b128(vt + vsrc1 + kc + 32, &vtile[p ^ 1][ku1 * 8]);
        }

        // ---- scores: two 16-key tiles x two 32-wide d chunks ----
        v8f s0 = (v8f){}, s1 = (v8f){};
#pragma unroll
        for (int c = 0; c < 2; c++) {
            v16bf bk0 = load_frag_b(&kt[p][(r)      * 64 + c * 32 + half * 16]);
            v16bf bk1 = load_frag_b(&kt[p][(16 + r) * 64 + c * 32 + half * 16]);
            s0 = wmma_bf16(aq[c], bk0, s0);
            s1 = wmma_bf16(aq[c], bk1, s1);
        }

        // ---- online softmax (exp2 domain); only the max needs shuffles ----
#pragma unroll
        for (int v = 0; v < 8; v++) {
            float x0 = s0[v], x1 = s1[v];
            float mx = fmaxf(x0, x1);
            mx = fmaxf(mx, __shfl_xor(mx, 1, 32));
            mx = fmaxf(mx, __shfl_xor(mx, 2, 32));
            mx = fmaxf(mx, __shfl_xor(mx, 4, 32));
            mx = fmaxf(mx, __shfl_xor(mx, 8, 32));
            float mnew  = fmaxf(row_m[v], mx);
            float alpha = __builtin_amdgcn_exp2f(row_m[v] - mnew);
            row_m[v] = mnew;
            x0 = __builtin_amdgcn_exp2f(x0 - mnew);
            x1 = __builtin_amdgcn_exp2f(x1 - mnew);
            int Mrow = v + 8 * half;
            pl[Mrow * 32 + r]      = (__bf16)x0;
            pl[Mrow * 32 + 16 + r] = (__bf16)x1;
            lacc[v]    *= alpha;
            cacc[0][v] *= alpha;
            cacc[1][v] *= alpha;
            cacc[2][v] *= alpha;
            cacc[3][v] *= alpha;
        }
        asm volatile("s_wait_dscnt 0" ::: "memory");
        v16bf pa = load_frag_a(pl + r * 32 + half * 16);

        // ---- row sums via the matrix pipe (broadcast to all lanes) ----
        lacc = wmma_bf16(pa, vones, lacc);

        // ---- ctx += P * V ----
#pragma unroll
        for (int t = 0; t < 4; t++) {
            v16bf bv = load_frag_b(&vtile[p][(16 * t + r) * 32 + half * 16]);
            cacc[t] = wmma_bf16(pa, bv, cacc[t]);
        }

        p ^= 1;
    }

    // ---- normalize and store ctx bf16 [B*S, 1024] ----
#pragma unroll
    for (int v = 0; v < 8; v++) {
        float inv = __builtin_amdgcn_rcpf(lacc[v]);
        int s = q0 + v + 8 * half;
#pragma unroll
        for (int t = 0; t < 4; t++) {
            int dd = h * 64 + 16 * t + r;
            ctx[(b * S + s) * 1024 + dd] = (__bf16)(cacc[t][v] * inv);
        }
    }
}

// ---------------------------------------------------------------------------
// Host launch
// ---------------------------------------------------------------------------
extern "C" void kernel_launch(void* const* d_in, const int* in_sizes, int n_in,
                              void* d_out, int out_size, void* d_ws, size_t ws_size,
                              hipStream_t stream) {
    (void)in_sizes; (void)n_in; (void)out_size; (void)ws_size;
    const int B = 4, S = 2048, D = 1024;
    const int M = B * S;          // 8192
    const int N3 = 3 * D;         // 3072
    const float QSCALE = 0.18033688011112042f;   // (1/sqrt(64)) * log2(e)

    const float* hidden = (const float*)d_in[0];
    const float* w_qkv  = (const float*)d_in[1];
    const float* w_out  = (const float*)d_in[2];
    float* out = (float*)d_out;

    char* ws = (char*)d_ws;
    __bf16* h_bf   = (__bf16*)ws;                 ws += (size_t)M * D * 2;
    __bf16* wqT    = (__bf16*)ws;                 ws += (size_t)N3 * D * 2;
    __bf16* woT    = (__bf16*)ws;                 ws += (size_t)D * D * 2;
    __bf16* qkv_bf = (__bf16*)ws;                 ws += (size_t)M * N3 * 2;
    __bf16* vt     = (__bf16*)ws;                 ws += (size_t)M * D * 2;
    __bf16* ctx_bf = h_bf;   // reuse: hidden dead after QKV GEMM

    {
        int n = M * D;
        cvt_f32_bf16<<<(n + 255) / 256, 256, 0, stream>>>(hidden, h_bf, n);
    }
    transpose_cvt<<<dim3(N3 / 256, D), 256, 0, stream>>>(w_qkv, wqT, D, N3);
    transpose_cvt<<<dim3(D / 256, D), 256, 0, stream>>>(w_out, woT, D, D);

    // QKV projection (Q columns pre-scaled into the exp2 domain)
    gemm_wmma<true><<<dim3(N3 / 128, M / 128), 256, 0, stream>>>(
        h_bf, wqT, nullptr, qkv_bf, M, N3, D, D, QSCALE);

    {
        int n = M * D;
        transpose_v<<<(n + 255) / 256, 256, 0, stream>>>(qkv_bf, vt);
    }

    attn_wmma<<<B * 16 * (S / 64), 128, 0, stream>>>(qkv_bf, vt, ctx_bf);

    // output projection -> f32
    gemm_wmma<false><<<dim3(D / 128, M / 128), 256, 0, stream>>>(
        ctx_bf, woT, out, nullptr, M, D, D, 0, 1.0f);
}
